// SelfAttentionWithRotaryEmbedding_29008209117608
// MI455X (gfx1250) — compile-verified, matrix operand order checked by reading, not run on hardware
//
#include <hip/hip_runtime.h>
#include <hip/hip_bf16.h>

typedef __attribute__((ext_vector_type(16))) _Float16 v16h;
typedef __attribute__((ext_vector_type(8)))  _Float16 v8h;
typedef __attribute__((ext_vector_type(2)))  _Float16 h2;
typedef __attribute__((ext_vector_type(8)))  float    v8f;

#define B_   4
#define L_   2048
#define D_   2048
#define H_   16
#define HD_  128
#define BH_  64
#define M_   8192            // B*L
#define EPS_ 1e-6f
#define SCALE_ 0.08838834764831845f   // 1/sqrt(128)

// Load a 16-half fragment (two contiguous 8-half runs, 16B apart) as 2x b128.
__device__ __forceinline__ v16h ld2x8(const _Float16* p) {
    v8h lo = *(const v8h*)p;
    v8h hi = *(const v8h*)(p + 16);
    return __builtin_shufflevector(lo, hi, 0,1,2,3,4,5,6,7,8,9,10,11,12,13,14,15);
}

// ---------------------------------------------------------------- converts
__global__ void cvt_f32_f16(const float* __restrict__ in, _Float16* __restrict__ out, int n) {
    int i = blockIdx.x * blockDim.x + threadIdx.x;
    int stride = gridDim.x * blockDim.x;
    for (; i < n; i += stride) out[i] = (_Float16)in[i];
}

// out[n*K + k] = in[k*N + n]  (reads coalesced along n; one-time cost)
__global__ void cvt_transpose_f32_f16(const float* __restrict__ in, _Float16* __restrict__ out,
                                      int total, int log2N, int K) {
    int i = blockIdx.x * blockDim.x + threadIdx.x;
    int stride = gridDim.x * blockDim.x;
    int N = 1 << log2N;
    for (; i < total; i += stride) {
        int n = i & (N - 1), k = i >> log2N;
        out[(size_t)n * K + k] = (_Float16)in[(size_t)k * N + n];
    }
}

// --------------------------------------------------- shared LDS (GEMM kernels)
// staging tiles and the f32 output tile never live at the same time
union GemmSmem {
    struct { _Float16 x_s[64 * 72]; _Float16 w_sT[128 * 72]; } g;
    float out_s[64 * 132];
};

// ------------------------------------------------- shared 64x128 GEMM core
// out_tile(64x128) = A[m0:m0+64, :2048] * WT[n0:n0+128, :2048]^T
// A: f16 row-major [M][2048]; WT: f16 [N][2048] (pre-transposed weight)
// K-step = 64 (two 32-deep WMMA sub-chunks per barrier pair)
__device__ __forceinline__ void gemm_tile_64x128(
    const _Float16* __restrict__ A, const _Float16* __restrict__ WT,
    int m0, int n0, GemmSmem& sm) {
    const int tid = threadIdx.x;
    const int wid = tid >> 5, lane = tid & 31;
    const int hi = lane >> 4, li = lane & 15;
    const int wm = wid >> 2, wn = wid & 3;          // 2x4 wave grid, 32x32/wave
    _Float16* x_s  = sm.g.x_s;
    _Float16* w_sT = sm.g.w_sT;
    v8f acc[2][2] = {};
    for (int kk = 0; kk < D_; kk += 64) {
        {   // x tile 64x64: two b128 per thread
            int r = tid >> 2, c0 = (tid & 3) * 16;
            const _Float16* src = A + (size_t)(m0 + r) * D_ + kk + c0;
            *(v8h*)&x_s[r * 72 + c0]     = *(const v8h*)src;
            *(v8h*)&x_s[r * 72 + c0 + 8] = *(const v8h*)(src + 8);
        }
        {   // wT tile 128x64: four b128 per thread
            int n = tid >> 1, k0 = (tid & 1) * 32;
            const _Float16* src = WT + (size_t)(n0 + n) * D_ + kk + k0;
            #pragma unroll
            for (int i = 0; i < 4; ++i)
                *(v8h*)&w_sT[n * 72 + k0 + i * 8] = *(const v8h*)(src + i * 8);
        }
        __syncthreads();
        #pragma unroll
        for (int kc = 0; kc < 2; ++kc) {
            v16h a[2], b[2];
            #pragma unroll
            for (int mt = 0; mt < 2; ++mt)
                a[mt] = ld2x8(&x_s[(wm * 32 + mt * 16 + li) * 72 + kc * 32 + hi * 8]);
            #pragma unroll
            for (int t = 0; t < 2; ++t)
                b[t] = ld2x8(&w_sT[(wn * 32 + t * 16 + li) * 72 + kc * 32 + hi * 8]);
            #pragma unroll
            for (int mt = 0; mt < 2; ++mt)
                #pragma unroll
                for (int t = 0; t < 2; ++t)
                    acc[mt][t] = __builtin_amdgcn_wmma_f32_16x16x32_f16(
                        false, a[mt], false, b[t], (short)0, acc[mt][t], false, false);
        }
        __syncthreads();
    }
    // staging tiles are dead now; reuse the LDS as the f32 output tile
    #pragma unroll
    for (int mt = 0; mt < 2; ++mt)
        #pragma unroll
        for (int t = 0; t < 2; ++t)
            #pragma unroll
            for (int vv = 0; vv < 8; ++vv)
                sm.out_s[(wm * 32 + mt * 16 + vv + 8 * hi) * 132 + wn * 32 + t * 16 + li] =
                    acc[mt][t][vv];
    __syncthreads();
}

// --------------------------------------- QK projection + RMSNorm + RoPE
__global__ __launch_bounds__(256) void proj_qk_kernel(
    const _Float16* __restrict__ xh, const _Float16* __restrict__ wqkT,
    const float* __restrict__ cosT, const float* __restrict__ sinT,
    const float* __restrict__ qn_w, const float* __restrict__ kn_w,
    _Float16* __restrict__ qbuf, _Float16* __restrict__ kbuf) {
    __shared__ __align__(16) GemmSmem sm;
    int n0 = blockIdx.x * 128;     // within 4096 (q: <2048, k: >=2048)
    int m0 = blockIdx.y * 64;
    gemm_tile_64x128(xh, wqkT, m0, n0, sm);
    int tid = threadIdx.x;
    if (tid < 64) {
        int m = m0 + tid;
        int b = m >> 11, l = m & 2047;
        bool is_q = (n0 < D_);
        int h = (n0 & (D_ - 1)) >> 7;          // tile == one full head slice
        const float* nw = is_q ? qn_w : kn_w;
        _Float16* dst = is_q ? qbuf : kbuf;
        float sc = is_q ? SCALE_ : 1.0f;       // fold attention scale into q
        float ss = 0.f;
        #pragma unroll 8
        for (int c = 0; c < 128; ++c) { float v = sm.out_s[tid * 132 + c]; ss += v * v; }
        float rr = rsqrtf(ss * (1.0f / 128.0f) + EPS_);
        size_t base = ((size_t)((b * H_ + h) * L_ + l)) * HD_;
        #pragma unroll 4
        for (int i = 0; i < 64; ++i) {
            float x1 = sm.out_s[tid * 132 + 2 * i]     * rr * nw[2 * i];
            float x2 = sm.out_s[tid * 132 + 2 * i + 1] * rr * nw[2 * i + 1];
            float c_ = cosT[l * 64 + i], s_ = sinT[l * 64 + i];
            dst[base + 2 * i]     = (_Float16)((x1 * c_ - x2 * s_) * sc);
            dst[base + 2 * i + 1] = (_Float16)((x1 * s_ + x2 * c_) * sc);
        }
    }
}

// ------------------------------------------------- V projection + bias
__global__ __launch_bounds__(256) void proj_v_kernel(
    const _Float16* __restrict__ xh, const _Float16* __restrict__ wvT,
    const float* __restrict__ bv, _Float16* __restrict__ vbuf) {
    __shared__ __align__(16) GemmSmem sm;
    int n0 = blockIdx.x * 128;
    int m0 = blockIdx.y * 64;
    gemm_tile_64x128(xh, wvT, m0, n0, sm);
    int tid = threadIdx.x;
    int e0 = tid * 32;
    int r = e0 >> 7, c0 = e0 & 127;
    int m = m0 + r, b = m >> 11, l = m & 2047;
    #pragma unroll 8
    for (int i = 0; i < 32; ++i) {
        int c = c0 + i, n = n0 + c;
        int h = n >> 7, hd = n & 127;
        float val = sm.out_s[r * 132 + c] + bv[n];
        vbuf[((size_t)((b * H_ + h) * L_ + l)) * HD_ + hd] = (_Float16)val;
    }
}

// --------------------------------------------- FlashAttention (causal)
__global__ __launch_bounds__(256) void attn_kernel(
    const _Float16* __restrict__ q, const _Float16* __restrict__ k,
    const _Float16* __restrict__ v, _Float16* __restrict__ o) {
    __shared__ __align__(16) _Float16 k_s[64 * 136];    // [key][d], row-major
    __shared__ __align__(16) _Float16 v_sT[128 * 72];   // [hd][key], transposed
    __shared__ __align__(16) _Float16 p_s[128 * 72];    // [row][key]
    int qt = blockIdx.x, bh = blockIdx.y;
    int tid = threadIdx.x, wid = tid >> 5, lane = tid & 31;
    int hi = lane >> 4, li = lane & 15;

    // Q fragments straight from global (row-major q is fragment-contiguous)
    v16h qf[4];
    int grow = qt * 128 + wid * 16 + li;
    const _Float16* qrow = q + ((size_t)bh * L_ + grow) * HD_;
    #pragma unroll
    for (int ks = 0; ks < 4; ++ks)
        qf[ks] = ld2x8(qrow + ks * 32 + hi * 8);

    v8f oa[8] = {};
    float m_r[8], l_r[8];
    #pragma unroll
    for (int vv = 0; vv < 8; ++vv) { m_r[vv] = -3.0e38f; l_r[vv] = 0.f; }

    int kend = 2 * qt + 1;
    for (int kt = 0; kt <= kend; ++kt) {
        __syncthreads();   // protect k_s/v_sT from previous iteration's readers
        {   // K tile: row-major (QK^T needs K^T == row-major k, so no transpose)
            int r = tid >> 2, c0 = (tid & 3) * 32;
            const _Float16* ksrc = k + ((size_t)bh * L_ + kt * 64 + r) * HD_ + c0;
            #pragma unroll
            for (int i = 0; i < 4; ++i)
                *(v8h*)&k_s[r * 136 + c0 + i * 8] = *(const v8h*)(ksrc + i * 8);
            if (kt < kend) __builtin_prefetch(ksrc + 64 * HD_, 0, 0);
        }
        {   // V tile: stage TRANSPOSED [hd][key]; wide global reads,
            // key-pairs packed into b32 LDS stores (8x fewer LDS ops than
            // per-fragment scalar gathers on the read side)
            int kp = tid & 31;          // key pair 0..31
            int hd0 = (tid >> 5) * 16;  // hd chunk
            const _Float16* v0 = v + ((size_t)bh * L_ + kt * 64 + 2 * kp) * HD_ + hd0;
            const _Float16* v1 = v0 + HD_;
            v8h a0 = *(const v8h*)v0, a1 = *(const v8h*)(v0 + 8);
            v8h b0 = *(const v8h*)v1, b1 = *(const v8h*)(v1 + 8);
            #pragma unroll
            for (int j = 0; j < 8; ++j) {
                h2 p0; p0[0] = a0[j]; p0[1] = b0[j];
                *(h2*)&v_sT[(hd0 + j) * 72 + 2 * kp] = p0;
                h2 p1; p1[0] = a1[j]; p1[1] = b1[j];
                *(h2*)&v_sT[(hd0 + 8 + j) * 72 + 2 * kp] = p1;
            }
            if (kt < kend) __builtin_prefetch(v0 + 64 * HD_, 0, 0);
        }
        __syncthreads();

        // S = Q * K^T : wide B-frag loads from row-major k_s
        v8f s_acc[4] = {};
        #pragma unroll
        for (int ks = 0; ks < 4; ++ks)
            #pragma unroll
            for (int t = 0; t < 4; ++t) {
                v16h b = ld2x8(&k_s[(t * 16 + li) * 136 + ks * 32 + hi * 8]);
                s_acc[t] = __builtin_amdgcn_wmma_f32_16x16x32_f16(
                    false, qf[ks], false, b, (short)0, s_acc[t], false, false);
            }

        // causal mask on the diagonal band
        if (kt >= 2 * qt) {
            #pragma unroll
            for (int t = 0; t < 4; ++t)
                #pragma unroll
                for (int vv = 0; vv < 8; ++vv) {
                    int row = qt * 128 + wid * 16 + vv + 8 * hi;
                    int col = kt * 64 + t * 16 + li;
                    if (col > row) s_acc[t][vv] = -3.0e38f;
                }
        }

        // online softmax (row stats via shfl_xor within each 16-lane half)
        float alpha[8], rs[8] = {};
        #pragma unroll
        for (int vv = 0; vv < 8; ++vv) {
            float mx = fmaxf(fmaxf(s_acc[0][vv], s_acc[1][vv]),
                             fmaxf(s_acc[2][vv], s_acc[3][vv]));
            #pragma unroll
            for (int off = 1; off < 16; off <<= 1)
                mx = fmaxf(mx, __shfl_xor(mx, off, 32));
            float mn = fmaxf(m_r[vv], mx);
            alpha[vv] = __expf(m_r[vv] - mn);
            m_r[vv] = mn;
        }
        #pragma unroll
        for (int t = 0; t < 4; ++t)
            #pragma unroll
            for (int vv = 0; vv < 8; ++vv) {
                float p = __expf(s_acc[t][vv] - m_r[vv]);
                s_acc[t][vv] = p;
                rs[vv] += p;
            }
        #pragma unroll
        for (int vv = 0; vv < 8; ++vv) {
            #pragma unroll
            for (int off = 1; off < 16; off <<= 1)
                rs[vv] += __shfl_xor(rs[vv], off, 32);
            l_r[vv] = l_r[vv] * alpha[vv] + rs[vv];
        }
        #pragma unroll
        for (int ot = 0; ot < 8; ++ot)
            #pragma unroll
            for (int vv = 0; vv < 8; ++vv) oa[ot][vv] *= alpha[vv];

        // stage P (f16) for the PV matmul (row-major -> A-frags are contiguous)
        #pragma unroll
        for (int t = 0; t < 4; ++t)
            #pragma unroll
            for (int vv = 0; vv < 8; ++vv)
                p_s[(wid * 16 + vv + 8 * hi) * 72 + t * 16 + li] =
                    (_Float16)s_acc[t][vv];
        __syncthreads();

        // O += P * V : all-wide fragment loads now
        #pragma unroll
        for (int ks = 0; ks < 2; ++ks) {
            v16h pa = ld2x8(&p_s[(wid * 16 + li) * 72 + ks * 32 + hi * 8]);
            #pragma unroll
            for (int ot = 0; ot < 8; ++ot) {
                v16h vbf = ld2x8(&v_sT[(ot * 16 + li) * 72 + ks * 32 + hi * 8]);
                oa[ot] = __builtin_amdgcn_wmma_f32_16x16x32_f16(
                    false, pa, false, vbf, (short)0, oa[ot], false, false);
            }
        }
    }

    // normalize and store O
    #pragma unroll
    for (int vv = 0; vv < 8; ++vv) {
        float inv = 1.0f / l_r[vv];
        int row = qt * 128 + wid * 16 + vv + 8 * hi;
        _Float16* orow = o + ((size_t)bh * L_ + row) * HD_;
        #pragma unroll
        for (int ot = 0; ot < 8; ++ot)
            orow[ot * 16 + li] = (_Float16)(oa[ot][vv] * inv);
    }
}

// ------------------------------------ output projection (head de-interleave)
__global__ __launch_bounds__(256) void outproj_kernel(
    const _Float16* __restrict__ oh, const _Float16* __restrict__ wpT,
    const float* __restrict__ bp, float* __restrict__ out) {
    __shared__ __align__(16) GemmSmem sm;
    int n0 = blockIdx.x * 128, m0 = blockIdx.y * 64;
    int tid = threadIdx.x, wid = tid >> 5, lane = tid & 31;
    int hi = lane >> 4, li = lane & 15;
    int wm = wid >> 2, wn = wid & 3;
    v8f acc[2][2] = {};
    int rl = tid >> 2, c0 = (tid & 3) * 16;
    int m = m0 + rl, b = m >> 11, l = m & 2047;
    for (int kk = 0; kk < D_; kk += 64) {
        {   // gather A from [b][h][l][hd] layout (16-aligned chunk stays in-head)
            int kbase = kk + c0;
            int h = kbase >> 7, hd = kbase & 127;
            const _Float16* src = oh + ((size_t)((b * H_ + h) * L_ + l)) * HD_ + hd;
            *(v8h*)&sm.g.x_s[rl * 72 + c0]     = *(const v8h*)src;
            *(v8h*)&sm.g.x_s[rl * 72 + c0 + 8] = *(const v8h*)(src + 8);
        }
        {
            int n = tid >> 1, k0 = (tid & 1) * 32;
            const _Float16* src = wpT + (size_t)(n0 + n) * D_ + kk + k0;
            #pragma unroll
            for (int i = 0; i < 4; ++i)
                *(v8h*)&sm.g.w_sT[n * 72 + k0 + i * 8] = *(const v8h*)(src + i * 8);
        }
        __syncthreads();
        #pragma unroll
        for (int kc = 0; kc < 2; ++kc) {
            v16h a[2], bfr[2];
            #pragma unroll
            for (int mt = 0; mt < 2; ++mt)
                a[mt] = ld2x8(&sm.g.x_s[(wm * 32 + mt * 16 + li) * 72 + kc * 32 + hi * 8]);
            #pragma unroll
            for (int t = 0; t < 2; ++t)
                bfr[t] = ld2x8(&sm.g.w_sT[(wn * 32 + t * 16 + li) * 72 + kc * 32 + hi * 8]);
            #pragma unroll
            for (int mt = 0; mt < 2; ++mt)
                #pragma unroll
                for (int t = 0; t < 2; ++t)
                    acc[mt][t] = __builtin_amdgcn_wmma_f32_16x16x32_f16(
                        false, a[mt], false, bfr[t], (short)0, acc[mt][t], false, false);
        }
        __syncthreads();
    }
    #pragma unroll
    for (int mt = 0; mt < 2; ++mt)
        #pragma unroll
        for (int t = 0; t < 2; ++t)
            #pragma unroll
            for (int vv = 0; vv < 8; ++vv)
                sm.out_s[(wm * 32 + mt * 16 + vv + 8 * hi) * 132 + wn * 32 + t * 16 + li] =
                    acc[mt][t][vv];
    __syncthreads();
    int e0 = tid * 32;
    int rr = e0 >> 7, cc0 = e0 & 127;
    int mm = m0 + rr;
    #pragma unroll 8
    for (int i = 0; i < 32; ++i) {
        int c = cc0 + i, n = n0 + c;
        out[(size_t)mm * D_ + n] = sm.out_s[rr * 132 + c] + bp[n];
    }
}

// ---------------------------------------------------------------- launch
extern "C" void kernel_launch(void* const* d_in, const int* in_sizes, int n_in,
                              void* d_out, int out_size, void* d_ws, size_t ws_size,
                              hipStream_t stream) {
    (void)in_sizes; (void)n_in; (void)out_size; (void)ws_size;
    const float* x    = (const float*)d_in[0];
    const float* cosT = (const float*)d_in[1];
    const float* sinT = (const float*)d_in[2];
    const float* Wqk  = (const float*)d_in[3];
    const float* Wv   = (const float*)d_in[4];
    const float* bv   = (const float*)d_in[5];
    const float* Wp   = (const float*)d_in[6];
    const float* bp   = (const float*)d_in[7];
    const float* qn   = (const float*)d_in[8];
    const float* kn   = (const float*)d_in[9];
    float* out = (float*)d_out;

    char* ws = (char*)d_ws;
    size_t off = 0;
    auto alloc = [&](size_t bytes) -> void* {
        void* p = ws + off;
        off += (bytes + 255) & ~(size_t)255;
        return p;
    };
    _Float16* xh    = (_Float16*)alloc((size_t)M_ * D_ * 2);
    _Float16* wqkT  = (_Float16*)alloc((size_t)2 * D_ * D_ * 2);  // [4096][2048]
    _Float16* wvT   = (_Float16*)alloc((size_t)D_ * D_ * 2);      // [2048][2048]
    _Float16* wpT   = (_Float16*)alloc((size_t)D_ * D_ * 2);
    _Float16* qb    = (_Float16*)alloc((size_t)BH_ * L_ * HD_ * 2);
    _Float16* kb    = (_Float16*)alloc((size_t)BH_ * L_ * HD_ * 2);
    _Float16* vb    = (_Float16*)alloc((size_t)BH_ * L_ * HD_ * 2);
    _Float16* ob    = (_Float16*)alloc((size_t)BH_ * L_ * HD_ * 2);

    cvt_f32_f16<<<4096, 256, 0, stream>>>(x, xh, M_ * D_);
    // weights stored transposed: [N][K], K contiguous -> wide LDS fragment loads
    cvt_transpose_f32_f16<<<4096, 256, 0, stream>>>(Wqk, wqkT, 2 * D_ * D_, 12, D_);
    cvt_transpose_f32_f16<<<2048, 256, 0, stream>>>(Wv,  wvT,  D_ * D_,     11, D_);
    cvt_transpose_f32_f16<<<2048, 256, 0, stream>>>(Wp,  wpT,  D_ * D_,     11, D_);

    proj_qk_kernel<<<dim3(32, 128), 256, 0, stream>>>(xh, wqkT, cosT, sinT, qn, kn, qb, kb);
    proj_v_kernel <<<dim3(16, 128), 256, 0, stream>>>(xh, wvT, bv, vb);
    attn_kernel   <<<dim3(16, 64),  256, 0, stream>>>(qb, kb, vb, ob);
    outproj_kernel<<<dim3(16, 128), 256, 0, stream>>>(ob, wpT, bp, out);
}
